// ContrastiveGNN_89799176224886
// MI455X (gfx1250) — matrix operand order repaired and checked by reading, not compiled, not run on hardware
//
#include <hip/hip_runtime.h>

#define BATCH   8
#define NPTS    2048
#define BN      (BATCH * NPTS)       // 16384 nodes
#define KNN     16
#define EDGES   (BN * KNN)           // 262144 edges
#define HID     128
#define PROJ_D  256
#define NUM_MP  3

typedef __attribute__((ext_vector_type(16))) __bf16 v16bf;
typedef __attribute__((ext_vector_type(8)))  float  v8f;

union Frag32B { uint4 q[2]; v16bf v; };

__device__ __forceinline__ unsigned short f2bf(float x) {
    unsigned int u = __float_as_uint(x);
    unsigned int r = (u + 0x7FFFu + ((u >> 16) & 1u)) >> 16;   // round-to-nearest-even
    return (unsigned short)r;
}

// ---------------- utility kernels ----------------

__global__ void zero_u32_kernel(unsigned int* p, int n) {
    int i = blockIdx.x * blockDim.x + threadIdx.x;
    if (i < n) p[i] = 0u;
}

__global__ void conv_bf16_kernel(const float* __restrict__ s, unsigned short* __restrict__ d, int n) {
    int i = blockIdx.x * blockDim.x + threadIdx.x;
    if (i < n) d[i] = f2bf(s[i]);
}

__global__ void invdeg_kernel(const int* __restrict__ deg, float* __restrict__ inv) {
    int i = blockIdx.x * blockDim.x + threadIdx.x;
    if (i < BN) {
        int d = deg[i];
        inv[i] = 1.0f / (float)(d < 1 ? 1 : d);
    }
}

// agg_scaled_bf16 = bf16(agg * inv_deg[node])
__global__ void scale_conv_kernel(const float* __restrict__ agg, const float* __restrict__ inv,
                                  unsigned short* __restrict__ aggb) {
    int t = blockIdx.x * blockDim.x + threadIdx.x;   // BN*HID threads
    aggb[t] = f2bf(agg[t] * inv[t >> 7]);
}

// encoder layer 1: relu(x @ enc_w0 + enc_b0), 5 -> 128, output bf16
__global__ void enc1_kernel(const float* __restrict__ x, const float* __restrict__ w0,
                            const float* __restrict__ b0, unsigned short* __restrict__ h0b) {
    int t = blockIdx.x * blockDim.x + threadIdx.x;   // BN*HID threads
    int node = t >> 7, c = t & 127;
    const float* xr = x + (size_t)node * 5;
    float s = b0[c];
#pragma unroll
    for (int d = 0; d < 5; ++d) s += xr[d] * w0[d * HID + c];
    h0b[t] = f2bf(fmaxf(s, 0.0f));
}

// ---------------- periodic kNN (one thread per node, LDS-staged batch positions) ----------------

__global__ void __launch_bounds__(256) knn_kernel(const float* __restrict__ pos,
                                                  int* __restrict__ nbr, int* __restrict__ deg) {
    __shared__ float ps[NPTS * 3];                   // 24 KB
    int b = blockIdx.x >> 3;                         // 8 blocks per batch (2048/256)
    int i = ((blockIdx.x & 7) << 8) + threadIdx.x;   // node within batch
    const float* pb = pos + (size_t)b * NPTS * 3;
    for (int t = threadIdx.x; t < NPTS * 3; t += 256) ps[t] = pb[t];
    __syncthreads();

    float xi = ps[i * 3 + 0], yi = ps[i * 3 + 1], zi = ps[i * 3 + 2];
    float bd[KNN]; int bi[KNN];
#pragma unroll
    for (int q = 0; q < KNN; ++q) { bd[q] = 3.0e38f; bi[q] = 0; }

    for (int j = 0; j < NPTS; ++j) {
        float dx = fabsf(xi - ps[j * 3 + 0]); dx = dx > 0.5f ? 1.0f - dx : dx;
        float dy = fabsf(yi - ps[j * 3 + 1]); dy = dy > 0.5f ? 1.0f - dy : dy;
        float dz = fabsf(zi - ps[j * 3 + 2]); dz = dz > 0.5f ? 1.0f - dz : dz;
        float d2 = dx * dx + dy * dy + dz * dz;
        if (j != i && d2 < bd[KNN - 1]) {
            bd[KNN - 1] = d2; bi[KNN - 1] = j;
#pragma unroll
            for (int q = KNN - 1; q > 0; --q) {      // branch-free bubble-up
                if (bd[q] < bd[q - 1]) {
                    float td = bd[q]; bd[q] = bd[q - 1]; bd[q - 1] = td;
                    int   ti = bi[q]; bi[q] = bi[q - 1]; bi[q - 1] = ti;
                }
            }
        }
    }
    int node = b * NPTS + i;
#pragma unroll
    for (int q = 0; q < KNN; ++q) {
        int d = b * NPTS + bi[q];
        nbr[node * KNN + q] = d;
        atomicAdd(&deg[d], 1);
    }
}

// exclusive scan of deg (BN entries) into cursor — single block, 256 threads
__global__ void __launch_bounds__(256) scan_kernel(const int* __restrict__ deg,
                                                   int* __restrict__ cursor) {
    __shared__ int sums[256];
    const int PER = BN / 256;                        // 64
    int t = threadIdx.x;
    int base = t * PER;
    int s = 0;
    for (int i = 0; i < PER; ++i) s += deg[base + i];
    sums[t] = s;
    __syncthreads();
    if (t == 0) {
        int acc = 0;
        for (int i = 0; i < 256; ++i) { int v = sums[i]; sums[i] = acc; acc += v; }
    }
    __syncthreads();
    int off = sums[t];
    for (int i = 0; i < PER; ++i) { cursor[base + i] = off; off += deg[base + i]; }
}

// reverse CSR fill: edges regrouped by destination
__global__ void csr_fill_kernel(const int* __restrict__ nbr, int* __restrict__ cursor,
                                int* __restrict__ rsrc, int* __restrict__ rdst) {
    int e = blockIdx.x * blockDim.x + threadIdx.x;   // EDGES
    int src = e >> 4;                                // node-major, KNN per node
    int dst = nbr[e];
    int pos = atomicAdd(&cursor[dst], 1);
    rsrc[pos] = src;
    rdst[pos] = dst;
}

// ---------------- WMMA GEMM: MTILE rows x N cols per block (8 waves, S=MTILE/16 subtiles) ------
// MODE 0: plain   : A0[row, 0..127]                      -> outF/outB[row,n]  (+bias, opt relu)
// MODE 1: message : concat(hb[rsrc(e)], hb[rdst(e)])     -> segmented atomicAdd agg[rdst] (relu)
//                   (A tile staged with GLOBAL_LOAD_ASYNC_TO_LDS_B128, ASYNCcnt tracked)
// MODE 2: update  : concat(hb[node], aggb[node])         -> h += relu(.), h/hb rewritten

template <int MODE, int MTILE>
__global__ void __launch_bounds__(256) gemm_kernel(
    const unsigned short* __restrict__ A0,
    const unsigned short* __restrict__ A1,
    const int* __restrict__ rsrc,
    const int* __restrict__ rdst,
    const unsigned short* __restrict__ W,     // K x N bf16
    const float* __restrict__ bias,
    float* outF, unsigned short* outB,
    int N, int reluFlag)
{
    constexpr int K = (MODE == 0) ? 128 : 256;
    constexpr int S = MTILE / 16;
    __shared__ __align__(16) unsigned short As[MTILE * K];
    __shared__ int Se[MTILE], De[MTILE];
    const int r0 = blockIdx.x * MTILE;

    // ---- stage A tile (MTILE rows x K bf16) into LDS ----
    if (MODE == 1) {
        if (threadIdx.x < MTILE) {
            Se[threadIdx.x] = rsrc[r0 + threadIdx.x];
            De[threadIdx.x] = rdst[r0 + threadIdx.x];
        }
        __syncthreads();
        for (int t = threadIdx.x; t < MTILE * (K / 8); t += 256) {   // 16B chunks
            int row = t >> 5, c = t & 31;
            int node = (c < 16) ? Se[row] : De[row];
            const unsigned short* hr = A0 + (size_t)node * HID + (size_t)(c & 15) * 8;
            unsigned ldsOff = (unsigned)(size_t)(As + (size_t)t * 8);
            asm volatile("global_load_async_to_lds_b128 %0, %1, off"
                         :: "v"(ldsOff), "v"((unsigned long long)(size_t)hr)
                         : "memory");
        }
        asm volatile("s_wait_asynccnt 0x0" ::: "memory");
    } else if (MODE == 2) {
        for (int t = threadIdx.x; t < MTILE * (K / 8); t += 256) {
            int row = t >> 5, c = t & 31;
            int node = r0 + row;
            const unsigned short* hr = (c < 16) ? (A0 + (size_t)node * HID)
                                                : (A1 + (size_t)node * HID);
            ((uint4*)As)[t] = ((const uint4*)hr)[c & 15];
        }
    } else {
        for (int t = threadIdx.x; t < MTILE * (K / 8); t += 256) {
            int row = t >> 4, c = t & 15;
            ((uint4*)As)[t] = ((const uint4*)(A0 + (size_t)(r0 + row) * HID))[c];
        }
    }
    __syncthreads();

    const int wave = threadIdx.x >> 5;
    const int lane = threadIdx.x & 31;
    const int hi   = (lane & 16) ? 8 : 0;    // K-chunk offset for A, M offset for D
    const int l15  = lane & 15;              // M for A-frag, N for B/D-frag
    const int nTiles = N >> 4;
    const v8f vzero = {0.f, 0.f, 0.f, 0.f, 0.f, 0.f, 0.f, 0.f};

    for (int ct = wave; ct < nTiles; ct += 8) {
        const int n0 = ct << 4;
        v8f acc[S];
#pragma unroll
        for (int s = 0; s < S; ++s) acc[s] = vzero;

#pragma unroll
        for (int k0 = 0; k0 < K; k0 += 32) {
            // B 32x16 bf16: lane holds N=l15, K = {hi..hi+7} U {hi+16..hi+23} (rel. k0) (shared by all S)
            Frag32B b;
            const unsigned short* wp = W + (size_t)(k0 + hi) * N + n0 + l15;
#pragma unroll
            for (int j = 0; j < 16; ++j)
                b.v[j] = __builtin_bit_cast(__bf16, wp[(size_t)((j & 7) + ((j >> 3) << 4)) * N]);
#pragma unroll
            for (int s = 0; s < S; ++s) {
                // A 16x32 bf16: lane holds M=l15, K = {hi..hi+7} U {hi+16..hi+23} (rel. k0)
                Frag32B a;
                const uint4* ap = (const uint4*)(As + (size_t)(s * 16 + l15) * K + k0 + hi);
                a.q[0] = ap[0];
                a.q[1] = ap[2];
                acc[s] = __builtin_amdgcn_wmma_f32_16x16x32_bf16(
                    false, a.v, false, b.v, (short)0, acc[s], false, false);
            }
        }

        // ---- epilogue: D lane layout (M = r + hi, N = l15) ----
        const int n = n0 + l15;
        const float bn = bias[n];
#pragma unroll
        for (int s = 0; s < S; ++s) {
            if (MODE == 1) {
                // rows grouped by dst -> run-length segmented atomic accumulation
                float run = 0.f;
                int prev = De[s * 16 + hi];
#pragma unroll
                for (int r = 0; r < 8; ++r) {
                    float v = fmaxf(acc[s][r] + bn, 0.0f);
                    int d = De[s * 16 + hi + r];
                    if (d != prev) {
                        atomicAdd(&outF[(size_t)prev * HID + n], run);
                        run = 0.f; prev = d;
                    }
                    run += v;
                }
                atomicAdd(&outF[(size_t)prev * HID + n], run);
            } else if (MODE == 2) {
#pragma unroll
                for (int r = 0; r < 8; ++r) {
                    int m = s * 16 + hi + r;
                    size_t idx = (size_t)(r0 + m) * HID + n;
                    float v = acc[s][r] + bn;
                    if (reluFlag) v = fmaxf(v, 0.0f);
                    float hv = outF[idx] + v;                    // residual update
                    outF[idx] = hv;
                    outB[idx] = f2bf(hv);
                }
            } else {
#pragma unroll
                for (int r = 0; r < 8; ++r) {
                    int m = s * 16 + hi + r;
                    size_t idx = (size_t)(r0 + m) * N + n;
                    float v = acc[s][r] + bn;
                    if (reluFlag) v = fmaxf(v, 0.0f);
                    outF[idx] = v;
                    if (outB) outB[idx] = f2bf(v);
                }
            }
        }
    }
}

// ---------------- host orchestration ----------------

extern "C" void kernel_launch(void* const* d_in, const int* in_sizes, int n_in,
                              void* d_out, int out_size, void* d_ws, size_t ws_size,
                              hipStream_t stream) {
    (void)in_sizes; (void)n_in; (void)out_size; (void)ws_size;

    const float* x       = (const float*)d_in[0];
    const float* pos     = (const float*)d_in[1];
    /* d_in[2] = k (== 16, fixed by setup)  */
    const float* enc_w0  = (const float*)d_in[3];
    const float* enc_b0  = (const float*)d_in[4];
    const float* enc_w1  = (const float*)d_in[5];
    const float* enc_b1  = (const float*)d_in[6];
    const float* msg_w   = (const float*)d_in[7];
    const float* msg_b   = (const float*)d_in[8];
    const float* upd_w   = (const float*)d_in[9];
    const float* upd_b   = (const float*)d_in[10];
    const float* proj_w0 = (const float*)d_in[11];
    const float* proj_b0 = (const float*)d_in[12];
    const float* proj_w1 = (const float*)d_in[13];
    const float* proj_b1 = (const float*)d_in[14];

    float* z_out = (float*)d_out;                            // (BN, 256)
    float* h_out = (float*)d_out + (size_t)BN * PROJ_D;      // (BN, 128)

    // workspace carve-out (256 B aligned)
    char* ws = (char*)d_ws;
    size_t cur = 0;
    auto carve = [&](size_t bytes) -> char* {
        char* p = ws + cur;
        cur += (bytes + 255) & ~(size_t)255;
        return p;
    };
    int*            nbr   = (int*)           carve((size_t)EDGES * 4);
    int*            deg   = (int*)           carve((size_t)BN * 4);
    float*          invd  = (float*)         carve((size_t)BN * 4);
    int*            curs  = (int*)           carve((size_t)BN * 4);
    int*            rsrc  = (int*)           carve((size_t)EDGES * 4);
    int*            rdst  = (int*)           carve((size_t)EDGES * 4);
    float*          h     = (float*)         carve((size_t)BN * HID * 4);
    unsigned short* hb    = (unsigned short*)carve((size_t)BN * HID * 2);
    float*          agg   = (float*)         carve((size_t)BN * HID * 4);
    unsigned short* aggb  = (unsigned short*)carve((size_t)BN * HID * 2);
    unsigned short* wenc1 = (unsigned short*)carve((size_t)HID * HID * 2);
    unsigned short* wmsg  = (unsigned short*)carve((size_t)NUM_MP * 2 * HID * HID * 2);
    unsigned short* wupd  = (unsigned short*)carve((size_t)NUM_MP * 2 * HID * HID * 2);
    unsigned short* wp0   = (unsigned short*)carve((size_t)HID * HID * 2);
    unsigned short* wp1   = (unsigned short*)carve((size_t)HID * PROJ_D * 2);

    // weights -> bf16
    conv_bf16_kernel<<<(HID * HID + 255) / 256, 256, 0, stream>>>(enc_w1, wenc1, HID * HID);
    conv_bf16_kernel<<<(NUM_MP * 2 * HID * HID + 255) / 256, 256, 0, stream>>>(msg_w, wmsg, NUM_MP * 2 * HID * HID);
    conv_bf16_kernel<<<(NUM_MP * 2 * HID * HID + 255) / 256, 256, 0, stream>>>(upd_w, wupd, NUM_MP * 2 * HID * HID);
    conv_bf16_kernel<<<(HID * HID + 255) / 256, 256, 0, stream>>>(proj_w0, wp0, HID * HID);
    conv_bf16_kernel<<<(HID * PROJ_D + 255) / 256, 256, 0, stream>>>(proj_w1, wp1, HID * PROJ_D);

    // graph build + reverse CSR (edges grouped by destination)
    zero_u32_kernel<<<BN / 256, 256, 0, stream>>>((unsigned int*)deg, BN);
    knn_kernel<<<BN / 256, 256, 0, stream>>>(pos, nbr, deg);
    invdeg_kernel<<<BN / 256, 256, 0, stream>>>(deg, invd);
    scan_kernel<<<1, 256, 0, stream>>>(deg, curs);
    csr_fill_kernel<<<EDGES / 256, 256, 0, stream>>>(nbr, curs, rsrc, rdst);

    // encoder: L1 scalar (5->128, bf16 into aggb scratch), L2 WMMA (128->128)
    enc1_kernel<<<(BN * HID) / 256, 256, 0, stream>>>(x, enc_w0, enc_b0, aggb);
    gemm_kernel<0, 32><<<BN / 32, 256, 0, stream>>>(aggb, nullptr, nullptr, nullptr,
                                                    wenc1, enc_b1, h, hb, HID, 1);

    // message passing
    for (int i = 0; i < NUM_MP; ++i) {
        zero_u32_kernel<<<(BN * HID) / 256, 256, 0, stream>>>((unsigned int*)agg, BN * HID);
        gemm_kernel<1, 64><<<EDGES / 64, 256, 0, stream>>>(hb, nullptr, rsrc, rdst,
                                                           wmsg + (size_t)i * 2 * HID * HID,
                                                           msg_b + (size_t)i * HID,
                                                           agg, nullptr, HID, 1);
        scale_conv_kernel<<<(BN * HID) / 256, 256, 0, stream>>>(agg, invd, aggb);
        gemm_kernel<2, 32><<<BN / 32, 256, 0, stream>>>(hb, aggb, nullptr, nullptr,
                                                        wupd + (size_t)i * 2 * HID * HID,
                                                        upd_b + (size_t)i * HID,
                                                        h, hb, HID, 1);
    }

    // projection head: relu(h @ w0 + b0) @ w1 + b1
    gemm_kernel<0, 32><<<BN / 32, 256, 0, stream>>>(hb, nullptr, nullptr, nullptr,
                                                    wp0, proj_b0, agg, aggb, HID, 1);
    gemm_kernel<0, 32><<<BN / 32, 256, 0, stream>>>(aggb, nullptr, nullptr, nullptr,
                                                    wp1, proj_b1, z_out, nullptr, PROJ_D, 0);

    // second tuple output: final node features h (f32)
    hipMemcpyAsync(h_out, h, (size_t)BN * HID * 4, hipMemcpyDeviceToDevice, stream);
}
